// LSTM_52201032516069
// MI455X (gfx1250) — matrix operand order repaired and checked by reading, not compiled, not run on hardware
//
#include <hip/hip_runtime.h>
#include <stdint.h>

// Problem constants (match reference): B=128, T=512, D=256, H=256
#define B_    128
#define T_    512
#define D_    256
#define H_    256
#define KTOT  512           // D + H (combined GEMM depth)
#define N4H   1024          // 4*H gate columns
#define WLO   (N4H * KTOT)  // element offset of lo-part weight table

// LDS strides (elements)
#define XS    260           // fp32 x staging rows (16 x 256)
#define AS    264           // bf16 A-operand rows (16 x 256), 528B stride
#define MP    20            // gates column stride [n][m], 80B: b128-aligned

typedef __attribute__((ext_vector_type(4)))  float  v4f;
typedef __attribute__((ext_vector_type(8)))  float  v8f;
typedef __attribute__((ext_vector_type(16))) __bf16 v16bf;
typedef __attribute__((ext_vector_type(8)))  __bf16 v8bf;

// fast activations: v_exp_f32 + v_rcp_f32 (+fma); ~1ulp rcp error,
// far below the bf16x3 GEMM noise floor
__device__ __forceinline__ float sigm_(float x) {
    return __builtin_amdgcn_rcpf(1.0f + __expf(-x));
}
__device__ __forceinline__ float tanh_(float x) {
    float e = __expf(2.0f * x);      // safe: e->inf => 1, e->0 => -1
    return __builtin_fmaf(-2.0f, __builtin_amdgcn_rcpf(e + 1.0f), 1.0f);
}

// 16-bit A fragment: elems 0-15 = K[koff..+15] (contiguous, matches B)
__device__ __forceinline__ v16bf load_a(const __bf16* row, int k) {
    v8bf c0 = *(const v8bf*)(row + k);
    v8bf c1 = *(const v8bf*)(row + k + 8);
    return __builtin_shufflevector(c0, c1, 0, 1, 2, 3, 4, 5, 6, 7,
                                           8, 9, 10, 11, 12, 13, 14, 15);
}

// ---- one-shot: split fp32 weights into combined bf16 hi/lo table ----------
// layout: whi[n*512 + k], k<256 -> W_ih[n][k], k>=256 -> W_hh[n][k-256]
__global__ void split_weights(const float* __restrict__ Wih,
                              const float* __restrict__ Whh,
                              __bf16* __restrict__ whi) {
    int idx = blockIdx.x * blockDim.x + threadIdx.x;   // 0 .. 1024*512-1
    int n = idx >> 9;
    int k = idx & 511;
    float w  = (k < D_) ? Wih[n * D_ + k] : Whh[n * H_ + (k - D_)];
    __bf16 hi = (__bf16)w;
    whi[idx]       = hi;
    whi[idx + WLO] = (__bf16)(w - (float)hi);          // lo table at +1MB
}

// ---- fused persistent LSTM scan: one 16-row batch tile per workgroup ------
__launch_bounds__(256, 1)
__global__ void lstm_fused_wmma_bf16x3(const float* __restrict__ x,
                                       const int*   __restrict__ x_len,
                                       const __bf16* __restrict__ whi,
                                       const float* __restrict__ b_ih,
                                       const float* __restrict__ b_hh,
                                       float*       __restrict__ out) {
    // hot A-operand arrays first: low LDS offsets -> ds_load immediate folding
    __shared__ __bf16 xhi[16 * AS];   // split x tile
    __shared__ __bf16 xlo[16 * AS];
    __shared__ __bf16 hhi[16 * AS];   // split hidden state
    __shared__ __bf16 hlo[16 * AS];
    __shared__ float  gl2[N4H * MP];  // gates, column-major [n][m]
    __shared__ float  xl[2][16 * XS]; // fp32 x staging, double-buffered

    const int tid  = threadIdx.x;     // 256 threads = 8 wave32s
    const int lane = tid & 31;
    const int wv   = tid >> 5;
    const int b0   = blockIdx.x * 16;

    // pointwise setup: thread owns hidden column j for all 16 rows
    const int j = tid;                // 0..255 == H-1
    const float bsi = b_ih[j]          + b_hh[j];
    const float bsf = b_ih[H_ + j]     + b_hh[H_ + j];
    const float bsg = b_ih[2 * H_ + j] + b_hh[2 * H_ + j];
    const float bso = b_ih[3 * H_ + j] + b_hh[3 * H_ + j];
    int   len[16];
    float cP[16];
#pragma unroll
    for (int m = 0; m < 16; ++m) { len[m] = x_len[b0 + m]; cP[m] = 0.0f; }

    // zero initial hidden state
#pragma unroll
    for (int m = 0; m < 16; ++m) {
        hhi[m * AS + tid] = (__bf16)0.0f;
        hlo[m * AS + tid] = (__bf16)0.0f;
    }

    // WMMA lane mapping
    const int mA    = lane & 15;
    const int koffA = (lane >> 4) * 16;       // 16-bit A K sub-offset
    const int koffB = (lane >> 4) * 16;       // 16-bit B K sub-offset
    const int nlane = lane & 15;

    const __bf16* arow_xh = &xhi[mA * AS + koffA];
    const __bf16* arow_xl = &xlo[mA * AS + koffA];
    const __bf16* arow_hh = &hhi[mA * AS + koffA];
    const __bf16* arow_hl = &hlo[mA * AS + koffA];

    // per-wave weight row pointers: wave wv owns N-tiles wv*8 .. wv*8+7
    const __bf16* wb[8];
#pragma unroll
    for (int i = 0; i < 8; ++i) {
        int n = (wv * 8 + i) * 16 + nlane;
        wb[i] = whi + (size_t)n * KTOT + koffB;
    }

    // async stage of one x timestep tile into LDS staging buffer `buf`
    auto stage_async = [&](int t, int buf) {
#pragma unroll
        for (int m = 0; m < 16; ++m) {
            const float* gp = x + ((size_t)(b0 + m) * T_ + t) * D_ + tid;
            unsigned lo = (unsigned)(uintptr_t)&xl[buf][m * XS + tid];
            asm volatile("global_load_async_to_lds_b32 %0, %1, off"
                         :: "v"(lo), "v"(gp) : "memory");
        }
    };

    stage_async(0, 0);                // prefetch t=0
    __syncthreads();                  // (covers hhi/hlo zero-init too)

    for (int t = 0; t < T_; ++t) {
        const int cb = t & 1;

        asm volatile("s_wait_asynccnt 0" ::: "memory");
        __syncthreads();              // x(t) resident in xl[cb]

        if (t + 1 < T_) stage_async(t + 1, cb ^ 1);   // hide behind GEMM

        // split x(t) into bf16 hi/lo (pure LDS->LDS, addrspace(3) preserved)
#pragma unroll 4
        for (int m = 0; m < 16; ++m) {
            float v = xl[cb][m * XS + tid];
            __bf16 hi = (__bf16)v;
            xhi[m * AS + tid] = hi;
            xlo[m * AS + tid] = (__bf16)(v - (float)hi);
        }
        __syncthreads();

        // ---- gates = [x_t | h] @ [W_ih | W_hh]^T  (bf16x3, fp32 accum) ----
        // three i-sweeps per K-block so consecutive WMMAs hit different accs
        v8f acc[8] = {};
#pragma unroll 2
        for (int kb = 0; kb < 8; ++kb) {          // K = 0..255 (x part)
            v16bf ah = load_a(arow_xh, kb * 32);
            v16bf al = load_a(arow_xl, kb * 32);
            v16bf bh[8];
#pragma unroll
            for (int i = 0; i < 8; ++i) bh[i] = *(const v16bf*)(wb[i] + kb * 32);
#pragma unroll
            for (int i = 0; i < 8; ++i)
                acc[i] = __builtin_amdgcn_wmma_f32_16x16x32_bf16(
                    false, ah, false, bh[i], (short)0, acc[i], false, false);
#pragma unroll
            for (int i = 0; i < 8; ++i)
                acc[i] = __builtin_amdgcn_wmma_f32_16x16x32_bf16(
                    false, al, false, bh[i], (short)0, acc[i], false, false);
            v16bf bl[8];
#pragma unroll
            for (int i = 0; i < 8; ++i) bl[i] = *(const v16bf*)(wb[i] + kb * 32 + WLO);
#pragma unroll
            for (int i = 0; i < 8; ++i)
                acc[i] = __builtin_amdgcn_wmma_f32_16x16x32_bf16(
                    false, ah, false, bl[i], (short)0, acc[i], false, false);
        }
#pragma unroll 2
        for (int kb = 0; kb < 8; ++kb) {          // K = 256..511 (h part)
            v16bf ah = load_a(arow_hh, kb * 32);
            v16bf al = load_a(arow_hl, kb * 32);
            v16bf bh[8];
#pragma unroll
            for (int i = 0; i < 8; ++i) bh[i] = *(const v16bf*)(wb[i] + 256 + kb * 32);
#pragma unroll
            for (int i = 0; i < 8; ++i)
                acc[i] = __builtin_amdgcn_wmma_f32_16x16x32_bf16(
                    false, ah, false, bh[i], (short)0, acc[i], false, false);
#pragma unroll
            for (int i = 0; i < 8; ++i)
                acc[i] = __builtin_amdgcn_wmma_f32_16x16x32_bf16(
                    false, al, false, bh[i], (short)0, acc[i], false, false);
            v16bf bl[8];
#pragma unroll
            for (int i = 0; i < 8; ++i)
                bl[i] = *(const v16bf*)(wb[i] + 256 + kb * 32 + WLO);
#pragma unroll
            for (int i = 0; i < 8; ++i)
                acc[i] = __builtin_amdgcn_wmma_f32_16x16x32_bf16(
                    false, ah, false, bl[i], (short)0, acc[i], false, false);
        }

        // ---- spill C fragments column-major: 2 x ds_store_b128 per tile ----
        const int mrow = (lane >> 4) * 8;         // rows mrow..mrow+7
#pragma unroll
        for (int i = 0; i < 8; ++i) {
            int n = (wv * 8 + i) * 16 + nlane;
            float* p = &gl2[n * MP + mrow];
            *(v4f*)(p)     = __builtin_shufflevector(acc[i], acc[i], 0, 1, 2, 3);
            *(v4f*)(p + 4) = __builtin_shufflevector(acc[i], acc[i], 4, 5, 6, 7);
        }
        __syncthreads();

        // ---- pointwise LSTM cell: 4 x ds_load_b128 per gate ----
        v4f G[4][4];                               // [gate][m-quad]
#pragma unroll
        for (int g = 0; g < 4; ++g) {
            const float* base = &gl2[(g * 256 + j) * MP];
#pragma unroll
            for (int q = 0; q < 4; ++q) G[g][q] = *(const v4f*)(base + 4 * q);
        }
#pragma unroll
        for (int m = 0; m < 16; ++m) {
            float gi = sigm_(G[0][m >> 2][m & 3] + bsi);
            float gf = sigm_(G[1][m >> 2][m & 3] + bsf);
            float gg = tanh_(G[2][m >> 2][m & 3] + bsg);
            float go = sigm_(G[3][m >> 2][m & 3] + bso);
            float c  = gf * cP[m] + gi * gg;
            cP[m]    = c;
            float h  = go * tanh_(c);
            __bf16 hh = (__bf16)h;
            hhi[m * AS + j] = hh;
            hlo[m * AS + j] = (__bf16)(h - (float)hh);
            float val = (t < len[m]) ? h : 0.0f;
            __builtin_nontemporal_store(
                val, &out[((size_t)(b0 + m) * T_ + t) * H_ + j]);
        }
        __syncthreads();              // h(t) ready for next step's fragments
    }
}

extern "C" void kernel_launch(void* const* d_in, const int* in_sizes, int n_in,
                              void* d_out, int out_size, void* d_ws, size_t ws_size,
                              hipStream_t stream) {
    const float* x    = (const float*)d_in[0];
    const int*   xlen = (const int*)  d_in[1];
    const float* Wih  = (const float*)d_in[2];
    const float* Whh  = (const float*)d_in[3];
    const float* bih  = (const float*)d_in[4];
    const float* bhh  = (const float*)d_in[5];
    float* out = (float*)d_out;
    (void)in_sizes; (void)n_in; (void)out_size; (void)ws_size;

    __bf16* whi = (__bf16*)d_ws;      // hi table; lo table at +WLO (2MB total)

    hipLaunchKernelGGL(split_weights, dim3((N4H * KTOT) / 256), dim3(256),
                       0, stream, Wih, Whh, whi);

    hipLaunchKernelGGL(lstm_fused_wmma_bf16x3, dim3(B_ / 16), dim3(256),
                       0, stream, x, xlen, whi, bih, bhh, out);
}
